// BCSM_11836929868330
// MI455X (gfx1250) — matrix operand order repaired and checked by reading, not compiled, not run on hardware
//
#include <hip/hip_runtime.h>

// ---------------------------------------------------------------------------
// Types for CDNA5 WMMA
// ---------------------------------------------------------------------------
typedef __attribute__((ext_vector_type(16))) _Float16 v16h;
typedef __attribute__((ext_vector_type(8)))  float    v8f;
typedef __attribute__((ext_vector_type(4)))  float    f4v;

// Model dims
#define D_MODEL 256
#define D_INNER 512
#define D_STATE 16
#define DT_RANK 16
#define BT      40      // B*L = 4*10
#define SEQ_L   10
#define NB      4       // batch
#define NE      8       // encoders

// ---------------------------------------------------------------------------
// Generic WMMA tile GEMM:  D(M=16 x N) = A(16 x K) @ W(N x K)^T
//   A      : LDS, fp32, row-major with stride ldA (rows 0..15)
//   W      : global fp32 weights, row-major (N,K)  (i.e. out = a @ W.T)
//   emit   : callback(m, n, value) for each output element owned by this lane
// One 16x16 output tile per wave iteration; 8 waves stride the N tiles.
// EXEC stays all-ones: tile loop is wave-uniform, no divergence around WMMA.
// ---------------------------------------------------------------------------
template <int K, class Emit>
__device__ inline void gemm_wmma(const float* A, int ldA,
                                 const float* __restrict__ W, int N,
                                 int tid, Emit emit) {
  const int wave = tid >> 5;
  const int lane = tid & 31;
  const int sub  = lane & 15;   // A row / B,C col within tile
  const int hh   = lane >> 4;   // lane half
  const int ntiles = N >> 4;
  for (int nt = wave; nt < ntiles; nt += 8) {
    const int n0 = nt << 4;
    v8f c = {};
    const float* wrow = W + (size_t)(n0 + sub) * K;
    for (int k0 = 0; k0 < K; k0 += 32) {
      v16h a, b;
      // A fragment: two contiguous 8-float runs: k0+8*hh .. +7  and  k0+16+8*hh .. +7
#pragma unroll
      for (int j = 0; j < 8; ++j) {
        int ka = k0 + 8 * hh + j;
        int kb = k0 + 16 + 8 * hh + j;
        a[j]     = (ka < K) ? (_Float16)A[sub * ldA + ka] : (_Float16)0.f;
        a[8 + j] = (kb < K) ? (_Float16)A[sub * ldA + kb] : (_Float16)0.f;
      }
      // B fragment: two 8-weight runs of row n: k0+8*hh .. +7  and  k0+16+8*hh .. +7
#pragma unroll
      for (int j = 0; j < 16; ++j) {
        int kk = k0 + 8 * hh + j + ((j >> 3) << 3);
        b[j] = (kk < K) ? (_Float16)wrow[kk] : (_Float16)0.f;
      }
      c = __builtin_amdgcn_wmma_f32_16x16x32_f16(false, a, false, b,
                                                 (short)0, c, false, false);
    }
    // C/D layout: lane -> N = lane&15 ; reg r -> M = r + 8*(lane>>4)
#pragma unroll
    for (int r = 0; r < 8; ++r) emit(r + 8 * hh, n0 + sub, c[r]);
  }
}

__device__ inline float silu_f(float x) { return x / (1.f + __expf(-x)); }

// ---------------------------------------------------------------------------
// Kernel 1: spatial mean pool.  grid = (40, 32), block = 256 (8 waves).
// wave w of block (bt, cg) reduces channel c = cg*8 + w over HW elements.
// ---------------------------------------------------------------------------
__global__ __launch_bounds__(256) void pool_kernel(const float* __restrict__ x,
                                                   float* __restrict__ out,
                                                   int HW) {
  const int bt   = blockIdx.x;
  const int wv   = threadIdx.x >> 5;
  const int lane = threadIdx.x & 31;
  const int c    = blockIdx.y * 8 + wv;
  const float* p = x + ((size_t)bt * 256 + c) * HW;
  float s = 0.f;
  const int n4 = HW >> 2;
  const f4v* p4 = (const f4v*)p;
  for (int i = lane; i < n4; i += 32) { f4v v = p4[i]; s += v.x + v.y + v.z + v.w; }
  for (int i = (n4 << 2) + lane; i < HW; i += 32) s += p[i];
#pragma unroll
  for (int o = 16; o >= 1; o >>= 1) s += __shfl_xor(s, o);
  if (lane == 0) out[bt * 256 + c] = s / (float)HW;
}

// ---------------------------------------------------------------------------
// Kernel 2: fused Mamba encoder stack.  grid = (4 batches, 8 encoders).
// One workgroup (256 threads, 8 waves) processes one (encoder, batch) pair:
// all activations live in LDS, M padded to one 16-row WMMA tile.
// ---------------------------------------------------------------------------
#define LDX 260   // stride for 256-wide buffers (pad +4 -> bank spread)
#define LDI 516   // stride for 512-wide fp32 buffers
#define LDZ 520   // stride for 512-wide fp16 buffer
#define LDB 52    // stride for 48-wide dbl buffer

__global__ __launch_bounds__(256) void encoder_kernel(
    const float* __restrict__ audio,   // (4,10,256)
    const float* __restrict__ vis_w,   // (4,256,256)
    const float* __restrict__ vis_b,   // (4,256)
    const float* __restrict__ ln_g,    // (8,3,256)
    const float* __restrict__ ln_b,    // (8,3,256)
    const float* __restrict__ in_w,    // (8,3,1024,256)
    const float* __restrict__ conv_w,  // (8,3,512,4)
    const float* __restrict__ conv_b,  // (8,3,512)
    const float* __restrict__ xp_w,    // (8,3,48,512)
    const float* __restrict__ dt_w,    // (8,3,512,16)
    const float* __restrict__ dt_b,    // (8,3,512)
    const float* __restrict__ A_log,   // (8,3,512,16)
    const float* __restrict__ Dp,      // (8,3,512)
    const float* __restrict__ out_w,   // (8,3,256,512)
    const float* __restrict__ gate_w,  // (8,256,256)
    const float* __restrict__ gate_b,  // (8,256)
    const float* __restrict__ pooled,  // ws: (4,40,256)
    float* __restrict__ gates)         // ws: (8,40,256)
{
  __shared__ float    s_x [16 * LDX];  // residual stream
  __shared__ float    s_xn[16 * LDX];  // layernorm output (A operand)
  __shared__ float    s_xi[16 * LDI];  // conv input
  __shared__ _Float16 s_z [16 * LDZ];  // gate half of in-proj
  __shared__ float    s_xc[16 * LDI];  // conv+silu output
  __shared__ float    s_dl[16 * LDI];  // delta, then y*silu(z)
  __shared__ float    s_db[16 * LDB];  // dt | B | C

  const int tid = threadIdx.x;
  const int b   = blockIdx.x;
  const int e   = blockIdx.y;

  // ---- build input sequence into s_x (rows 10..15 padded) ----
  if (e < 4) {
    for (int idx = tid; idx < 16 * 256; idx += 256) {
      int r = idx >> 8, c = idx & 255;
      s_x[r * LDX + c] = (r < SEQ_L) ? audio[(b * SEQ_L + r) * 256 + c] : 0.f;
    }
  } else {
    const int i = e - 4;
    for (int idx = tid; idx < 16 * 256; idx += 256) {
      int r = idx >> 8, c = idx & 255;
      s_xn[r * LDX + c] = (r < SEQ_L) ? pooled[((size_t)i * BT + b * SEQ_L + r) * 256 + c] : 0.f;
    }
    __syncthreads();
    const float* vb = vis_b + i * 256;
    gemm_wmma<256>(s_xn, LDX, vis_w + (size_t)i * 256 * 256, 256, tid,
                   [&](int m, int n, float v) { s_x[m * LDX + n] = v + vb[n]; });
  }
  __syncthreads();

  // ---- 3 Mamba layers ----
  for (int layer = 0; layer < 3; ++layer) {
    const int eL = e * 3 + layer;

    // LayerNorm: 8 waves x 2 rows
    {
      const float* g  = ln_g + eL * 256;
      const float* bb = ln_b + eL * 256;
      const int lane = tid & 31, wv = tid >> 5;
      for (int rr = 0; rr < 2; ++rr) {
        const int r = wv * 2 + rr;
        float vals[8], s = 0.f, s2 = 0.f;
#pragma unroll
        for (int j = 0; j < 8; ++j) {
          float v = s_x[r * LDX + lane + j * 32];
          vals[j] = v; s += v; s2 += v * v;
        }
#pragma unroll
        for (int o = 16; o >= 1; o >>= 1) { s += __shfl_xor(s, o); s2 += __shfl_xor(s2, o); }
        const float m   = s * (1.f / 256.f);
        const float var = s2 * (1.f / 256.f) - m * m;
        const float inv = rsqrtf(var + 1e-5f);
#pragma unroll
        for (int j = 0; j < 8; ++j) {
          int c = lane + j * 32;
          s_xn[r * LDX + c] = (vals[j] - m) * inv * g[c] + bb[c];
        }
      }
    }
    __syncthreads();

    // GEMM1: [xi | z] = xn @ in_w^T   (N=1024, K=256)
    gemm_wmma<256>(s_xn, LDX, in_w + (size_t)eL * 1024 * 256, 1024, tid,
                   [&](int m, int n, float v) {
                     if (n < 512) s_xi[m * LDI + n] = v;
                     else         s_z[m * LDZ + (n - 512)] = (_Float16)v;
                   });
    __syncthreads();

    // causal depthwise conv (taps 4) + silu
    {
      const float* cw = conv_w + (size_t)eL * 512 * 4;
      const float* cb = conv_b + eL * 512;
      for (int idx = tid; idx < 16 * 512; idx += 256) {
        const int r = idx >> 9, c = idx & 511;
        float v = 0.f;
        if (r < SEQ_L) {
          float acc = cb[c];
#pragma unroll
          for (int k = 0; k < 4; ++k) {
            int l = r - 3 + k;
            if (l >= 0) acc = fmaf(s_xi[l * LDI + c], cw[c * 4 + k], acc);
          }
          v = silu_f(acc);
        }
        s_xc[r * LDI + c] = v;
      }
    }
    __syncthreads();

    // GEMM2: dbl = xc @ xp_w^T   (N=48, K=512)
    gemm_wmma<512>(s_xc, LDI, xp_w + (size_t)eL * 48 * 512, 48, tid,
                   [&](int m, int n, float v) { s_db[m * LDB + n] = v; });
    __syncthreads();

    // GEMM3: delta = softplus(dt @ dt_w^T + dt_b)   (N=512, K=16 zero-padded)
    {
      const float* db = dt_b + eL * 512;
      gemm_wmma<16>(s_db, LDB, dt_w + (size_t)eL * 512 * 16, 512, tid,
                    [&](int m, int n, float v) {
                      float t = v + db[n];
                      s_dl[m * LDI + n] = (t > 20.f) ? t : log1pf(__expf(t));
                    });
    }
    __syncthreads();

    // selective scan: thread owns channels d = tid, tid+256 ; h[16] in VGPRs
    {
      const float* Al  = A_log + (size_t)eL * 512 * 16;
      const float* Dpp = Dp + eL * 512;
      for (int pass = 0; pass < 2; ++pass) {
        const int d = tid + pass * 256;
        float An[16], h[16];
#pragma unroll
        for (int n = 0; n < 16; ++n) { An[n] = -__expf(Al[d * 16 + n]); h[n] = 0.f; }
        const float Dv = Dpp[d];
        for (int l = 0; l < SEQ_L; ++l) {
          const float dl  = s_dl[l * LDI + d];
          const float xcv = s_xc[l * LDI + d];
          const float du  = dl * xcv;
          float acc = 0.f;
#pragma unroll
          for (int n = 0; n < 16; ++n) {
            float dA = __expf(dl * An[n]);
            h[n] = fmaf(dA, h[n], du * s_db[l * LDB + 16 + n]);   // B
            acc  = fmaf(h[n], s_db[l * LDB + 32 + n], acc);       // C
          }
          const float zv = (float)s_z[l * LDZ + d];
          s_dl[l * LDI + d] = (acc + xcv * Dv) * silu_f(zv);      // y*silu(z)
        }
      }
    }
    __syncthreads();

    // GEMM4: out = y @ out_w^T ; residual: x = xn + out   (N=256, K=512)
    gemm_wmma<512>(s_dl, LDI, out_w + (size_t)eL * 256 * 512, 256, tid,
                   [&](int m, int n, float v) { s_x[m * LDX + n] = s_xn[m * LDX + n] + v; });
    __syncthreads();
  }

  // gate head: silu(x @ gate_w^T + gate_b) -> global gates[e][b*10+m][n]
  {
    const float* gb = gate_b + e * 256;
    gemm_wmma<256>(s_x, LDX, gate_w + (size_t)e * 256 * 256, 256, tid,
                   [&](int m, int n, float v) {
                     if (m < SEQ_L) {
                       float t = v + gb[n];
                       gates[((size_t)e * BT + b * SEQ_L + m) * 256 + n] = silu_f(t);
                     }
                   });
  }
}

// ---------------------------------------------------------------------------
// Kernel 3: apply gate elementwise to a big tensor (streamed, NT stores).
// grid = (40, 256); block 256 threads loops HW.
// transposed=0 : gate row = bt           (B-major, for x1)
// transposed=1 : gate row = (bt%4)*10 + bt/4   (L-major, for x2..x4)
// ---------------------------------------------------------------------------
__global__ __launch_bounds__(256) void apply_gate_kernel(
    const float* __restrict__ x, float* __restrict__ out,
    const float* __restrict__ g, int HW, int transposed) {
  const int bt = blockIdx.x;
  const int c  = blockIdx.y;
  const int gr = transposed ? ((bt & 3) * 10 + (bt >> 2)) : bt;
  const float scale = 1.f + g[gr * 256 + c];
  const float* xp = x + ((size_t)bt * 256 + c) * HW;
  float*       op = out + ((size_t)bt * 256 + c) * HW;
  const int n4 = HW >> 2;
  const f4v* xp4 = (const f4v*)xp;
  f4v*       op4 = (f4v*)op;
  for (int i = threadIdx.x; i < n4; i += 256) {
    f4v v = xp4[i];
    v *= scale;
    __builtin_nontemporal_store(v, op4 + i);
  }
  for (int i = (n4 << 2) + threadIdx.x; i < HW; i += 256)
    __builtin_nontemporal_store(xp[i] * scale, op + i);
}

// audio output: ao = audio * (1 + mean(gates[4..7], transposed indexing))
__global__ __launch_bounds__(256) void audio_out_kernel(
    const float* __restrict__ audio, const float* __restrict__ gates,
    float* __restrict__ out) {
  const int bt = blockIdx.x;
  const int c  = threadIdx.x;
  const int gr = ((bt & 3) * 10 + (bt >> 2)) * 256 + c;
  const size_t G = (size_t)BT * 256;
  const float vg = 0.25f * (gates[4 * G + gr] + gates[5 * G + gr] +
                            gates[6 * G + gr] + gates[7 * G + gr]);
  out[bt * 256 + c] = audio[bt * 256 + c] * (1.f + vg);
}

// ---------------------------------------------------------------------------
extern "C" void kernel_launch(void* const* d_in, const int* in_sizes, int n_in,
                              void* d_out, int out_size, void* d_ws, size_t ws_size,
                              hipStream_t stream) {
  (void)in_sizes; (void)n_in; (void)out_size; (void)ws_size;
  const float* x1     = (const float*)d_in[0];
  const float* x2     = (const float*)d_in[1];
  const float* x3     = (const float*)d_in[2];
  const float* x4     = (const float*)d_in[3];
  const float* audio  = (const float*)d_in[4];
  const float* vis_w  = (const float*)d_in[5];
  const float* vis_b  = (const float*)d_in[6];
  const float* ln_g   = (const float*)d_in[7];
  const float* ln_b   = (const float*)d_in[8];
  const float* in_w   = (const float*)d_in[9];
  const float* conv_w = (const float*)d_in[10];
  const float* conv_b = (const float*)d_in[11];
  const float* xp_w   = (const float*)d_in[12];
  const float* dt_w   = (const float*)d_in[13];
  const float* dt_b   = (const float*)d_in[14];
  const float* A_log  = (const float*)d_in[15];
  const float* Dp     = (const float*)d_in[16];
  const float* out_w  = (const float*)d_in[17];
  const float* gate_w = (const float*)d_in[18];
  const float* gate_b = (const float*)d_in[19];

  float* ws     = (float*)d_ws;
  float* pooled = ws;                    // 4 * 40 * 256
  float* gates  = ws + 4 * BT * 256;     // 8 * 40 * 256

  float* out  = (float*)d_out;           // x4o | x3o | x2o | x1o | ao
  float* o_x4 = out;
  float* o_x3 = out + 501760;
  float* o_x2 = out + 2508800;
  float* o_x1 = out + 10536960;
  float* o_ao = out + 42649600;

  dim3 pg(BT, 32);
  pool_kernel<<<pg, 256, 0, stream>>>(x1, pooled + 0 * BT * 256, 56 * 56);
  pool_kernel<<<pg, 256, 0, stream>>>(x2, pooled + 1 * BT * 256, 28 * 28);
  pool_kernel<<<pg, 256, 0, stream>>>(x3, pooled + 2 * BT * 256, 14 * 14);
  pool_kernel<<<pg, 256, 0, stream>>>(x4, pooled + 3 * BT * 256, 7 * 7);

  encoder_kernel<<<dim3(NB, NE), 256, 0, stream>>>(
      audio, vis_w, vis_b, ln_g, ln_b, in_w, conv_w, conv_b, xp_w,
      dt_w, dt_b, A_log, Dp, out_w, gate_w, gate_b, pooled, gates);

  dim3 ag(BT, 256);
  apply_gate_kernel<<<ag, 256, 0, stream>>>(x1, o_x1, gates + 0 * BT * 256, 56 * 56, 0);
  apply_gate_kernel<<<ag, 256, 0, stream>>>(x2, o_x2, gates + 1 * BT * 256, 28 * 28, 1);
  apply_gate_kernel<<<ag, 256, 0, stream>>>(x3, o_x3, gates + 2 * BT * 256, 14 * 14, 1);
  apply_gate_kernel<<<ag, 256, 0, stream>>>(x4, o_x4, gates + 3 * BT * 256, 7 * 7, 1);
  audio_out_kernel<<<BT, 256, 0, stream>>>(audio, gates, o_ao);
}
